// MultiHeadAttention_6167573037102
// MI455X (gfx1250) — matrix operand order repaired and checked by reading, not compiled, not run on hardware
//
#include <hip/hip_runtime.h>

// ---------------------------------------------------------------------------
// CDNA5 (gfx1250) multi-head attention forward with KV cache.
// All matmuls use v_wmma_f32_16x16x32_bf16 with f32 accumulation.
// All hot-loop operands are pre-materialized as bf16 (L2-resident), so the
// GEMM/attention staging paths are pure b128 global->LDS copies.
// Pipeline registers are named scalars (NOT arrays/lambdas) so they stay in
// VGPRs instead of being promoted to LDS by the alloca-promotion pass.
// ---------------------------------------------------------------------------

typedef __bf16 bf16_t;
typedef __attribute__((ext_vector_type(16))) __bf16 v16bf;
typedef __attribute__((ext_vector_type(8)))  float  v8f;

union FragBF {
    v16bf v;
    uint4 q[2];
};

__device__ __forceinline__ bf16_t f2bf(float x) {
    unsigned u = __builtin_bit_cast(unsigned, x);
    unsigned r = u + 0x7FFFu + ((u >> 16) & 1u);   // round-to-nearest-even
    return __builtin_bit_cast(bf16_t, (unsigned short)(r >> 16));
}

__device__ __forceinline__ v8f wmma_bf16(v16bf a, v16bf b, v8f c) {
    return __builtin_amdgcn_wmma_f32_16x16x32_bf16(
        /*neg_a=*/false, a, /*neg_b=*/false, b,
        /*c_mod=*/(short)0, c, /*reuse_a=*/false, /*reuse_b=*/false);
}

// A fragment (16x32, MxK) from a row-major LDS tile, per ISA 16-bit A layout.
__device__ __forceinline__ v16bf load_a_frag(const bf16_t* t, int ld, int m0, int k0, int lane) {
    const int m  = m0 + (lane & 15);
    const int kb = k0 + ((lane >> 4) << 3);
    FragBF f;
    f.q[0] = *(const uint4*)(t + m * ld + kb);
    f.q[1] = *(const uint4*)(t + m * ld + kb + 16);
    return f.v;
}

// B fragment (32x16, KxN) where the LDS tile is stored [N][K] row-major.
__device__ __forceinline__ v16bf load_b_frag_nt(const bf16_t* t, int ld, int n0, int k0, int lane) {
    const int n  = n0 + (lane & 15);
    const int kb = k0 + ((lane >> 4) << 3);
    FragBF f;
    f.q[0] = *(const uint4*)(t + n * ld + kb);
    f.q[1] = *(const uint4*)(t + n * ld + kb + 16);
    return f.v;
}

// ---------------------------------------------------------------------------
// Bulk f32 -> bf16 conversion (one-time pass; output is L2-resident after).
// ---------------------------------------------------------------------------
__global__ __launch_bounds__(256)
void cvt_f32_to_bf16(const float4* __restrict__ src, uint2* __restrict__ dst, int n4) {
    const int i = blockIdx.x * 256 + threadIdx.x;
    if (i < n4) {
        const float4 v = src[i];
        union { bf16_t h[4]; uint2 u; } p;
        p.h[0] = f2bf(v.x); p.h[1] = f2bf(v.y);
        p.h[2] = f2bf(v.z); p.h[3] = f2bf(v.w);
        dst[i] = p.u;
    }
}

// ---------------------------------------------------------------------------
// GEMM: Out = X @ W^T + bias   (X, W pre-converted bf16 row-major, K=N=1024)
// mode 0: OutF[m,n] plain f32 [M,1024]                     (final projection)
// mode 1: OutB = bf16 Q head layout [B,H,2048,64], scaled by 0.125
// mode 2: OutF = f32 KV cache slot [B,H,4096,64] at t+2048, OutB = bf16 copy
// Software pipeline: global b128 loads for k-tile t+1 issued during WMMA of t,
// held in named VGPR scalars.
// ---------------------------------------------------------------------------
__global__ __launch_bounds__(256)
void mha_gemm_wmma(const bf16_t* __restrict__ X, const bf16_t* __restrict__ W,
                   const float* __restrict__ bias, float* __restrict__ OutF,
                   bf16_t* __restrict__ OutB, int mode) {
    constexpr int KDIM = 1024, NDIM = 1024, LD = 40;
    __shared__ bf16_t Xs[128 * LD];
    __shared__ bf16_t Wl[128 * LD];

    const int tid  = threadIdx.x;
    const int lane = tid & 31;
    const int w    = tid >> 5;   // 0..7
    const int wr   = w >> 1;     // 0..3
    const int wc   = w & 1;      // 0..1
    const int bm   = blockIdx.y * 128;
    const int bn   = blockIdx.x * 128;

    // Staging: thread covers one row (tid>>1), 16 halves at col (tid&1)*16.
    const int srow = tid >> 1;
    const int scol = (tid & 1) * 16;
    const bf16_t* __restrict__ xrow = X + (size_t)(bm + srow) * KDIM + scol;
    const bf16_t* __restrict__ wrow = W + (size_t)(bn + srow) * KDIM + scol;

    v8f acc[2][4];
#pragma unroll
    for (int i = 0; i < 2; ++i)
#pragma unroll
        for (int j = 0; j < 4; ++j)
            acc[i][j] = (v8f){0.f, 0.f, 0.f, 0.f, 0.f, 0.f, 0.f, 0.f};

    // Pipeline registers: named scalars (kept in VGPRs).
    uint4 gx0 = *(const uint4*)(xrow);
    uint4 gx1 = *(const uint4*)(xrow + 8);
    uint4 gw0 = *(const uint4*)(wrow);
    uint4 gw1 = *(const uint4*)(wrow + 8);

    for (int kt = 0; kt < KDIM; kt += 32) {
        __syncthreads();                       // LDS readers of prev tile done
        *(uint4*)(Xs + srow * LD + scol)     = gx0;
        *(uint4*)(Xs + srow * LD + scol + 8) = gx1;
        *(uint4*)(Wl + srow * LD + scol)     = gw0;
        *(uint4*)(Wl + srow * LD + scol + 8) = gw1;
        __syncthreads();

        const int ktn = kt + 32;
        if (ktn < KDIM) {                      // overlap next fetch with WMMA
            gx0 = *(const uint4*)(xrow + ktn);
            gx1 = *(const uint4*)(xrow + ktn + 8);
            gw0 = *(const uint4*)(wrow + ktn);
            gw1 = *(const uint4*)(wrow + ktn + 8);
        }

        const v16bf a0 = load_a_frag(Xs, LD, wr * 32,      0, lane);
        const v16bf a1 = load_a_frag(Xs, LD, wr * 32 + 16, 0, lane);
        v16bf bfr[4];
#pragma unroll
        for (int nt = 0; nt < 4; ++nt)
            bfr[nt] = load_b_frag_nt(Wl, LD, wc * 64 + nt * 16, 0, lane);
#pragma unroll
        for (int nt = 0; nt < 4; ++nt) {
            acc[0][nt] = wmma_bf16(a0, bfr[nt], acc[0][nt]);
            acc[1][nt] = wmma_bf16(a1, bfr[nt], acc[1][nt]);
        }
    }

    const int hi = lane >> 4;
    const int ln = lane & 15;
#pragma unroll
    for (int mf = 0; mf < 2; ++mf) {
#pragma unroll
        for (int nt = 0; nt < 4; ++nt) {
            const int n  = bn + wc * 64 + nt * 16 + ln;
            const float bv = bias[n];
#pragma unroll
            for (int r = 0; r < 8; ++r) {
                const int m = bm + wr * 32 + mf * 16 + r + hi * 8;
                const float val = acc[mf][nt][r] + bv;
                if (mode == 0) {
                    OutF[(size_t)m * NDIM + n] = val;
                } else {
                    const int bb = m >> 11, t = m & 2047;
                    const int hh = n >> 6,  d = n & 63;
                    if (mode == 1) {
                        const size_t off = (((size_t)(bb * 16 + hh)) * 2048 + t) * 64 + d;
                        OutB[off] = f2bf(val * 0.125f);   // fold 1/sqrt(64)
                    } else {
                        const size_t off = (((size_t)(bb * 16 + hh)) * 4096 + 2048 + t) * 64 + d;
                        OutF[off] = val;
                        OutB[off] = f2bf(val);
                    }
                }
            }
        }
    }
}

// ---------------------------------------------------------------------------
// Flash attention over the concatenated KV cache (all-bf16 operands).
// Qb: [B,H,2048,64] (pre-scaled), Kb/Vb: [B,H,4096,64], CtxB: [B,2048,1024].
// K staged [kv][d]; V staged TRANSPOSED [d][kv] (half-word merges).
// KV tile fetches pipelined one tile ahead in named VGPR scalars.
// ---------------------------------------------------------------------------
__global__ __launch_bounds__(256)
void mha_flash_attn_wmma(const bf16_t* __restrict__ Qb, const bf16_t* __restrict__ Kb,
                         const bf16_t* __restrict__ Vb, bf16_t* __restrict__ CtxB) {
    constexpr int LD  = 72;
    constexpr int LDV = 72;
    constexpr int T_PAST = 2048;
    constexpr float LOG2E = 1.44269504088896f;
    __shared__ bf16_t Qs[128 * LD];
    __shared__ bf16_t Ks[64 * LD];     // [kv][d]
    __shared__ bf16_t Vs[64 * LDV];    // [d][kv]  (transposed)
    __shared__ bf16_t Ps[128 * LD];

    const int tid  = threadIdx.x;
    const int lane = tid & 31;
    const int w    = tid >> 5;
    const int hi   = lane >> 4;
    const int ln   = lane & 15;
    const int q0   = blockIdx.x * 128;
    const int h    = blockIdx.y;
    const int b    = blockIdx.z;

    const size_t qbase  = (((size_t)(b * 16 + h)) * 2048 + q0) * 64;
    const size_t kvbase = ((size_t)(b * 16 + h)) * 4096 * 64;

    // Stage Q once: pure b128 copies (already bf16, already scaled)
    {
        const int row = tid >> 1;
        const int c0  = (tid & 1) * 32;
#pragma unroll
        for (int u = 0; u < 4; ++u)
            *(uint4*)(Qs + row * LD + c0 + u * 8) =
                *(const uint4*)(Qb + qbase + (size_t)row * 64 + c0 + u * 8);
    }

    // --- pipelined KV tile fetch state (named scalars -> VGPRs) ---
    // K: thread covers row tid>>2, 16 halves at col (tid&3)*16 (2x uint4).
    // V: thread covers kv row-pairs {vpr, vpr+16}, d cols vdc..vdc+3 (uint2 each).
    const int krow = tid >> 2;
    const int kcol = (tid & 3) * 16;
    const int vpr  = tid >> 4;          // 0..15 (second chunk: +16)
    const int vdc  = (tid & 15) * 4;
    const bf16_t* __restrict__ krp  = Kb + kvbase + (size_t)krow * 64 + kcol;
    const bf16_t* __restrict__ vrp0 = Vb + kvbase + (size_t)(2 * vpr) * 64 + vdc;
    const bf16_t* __restrict__ vrp1 = Vb + kvbase + (size_t)(2 * (vpr + 16)) * 64 + vdc;

    uint4 gk0 = *(const uint4*)(krp);
    uint4 gk1 = *(const uint4*)(krp + 8);
    uint2 gva0 = *(const uint2*)(vrp0);
    uint2 gvb0 = *(const uint2*)(vrp0 + 64);
    uint2 gva1 = *(const uint2*)(vrp1);
    uint2 gvb1 = *(const uint2*)(vrp1 + 64);

    v8f accO[4];
#pragma unroll
    for (int i = 0; i < 4; ++i)
        accO[i] = (v8f){0.f, 0.f, 0.f, 0.f, 0.f, 0.f, 0.f, 0.f};
    float mrun[8], lrun[8];
#pragma unroll
    for (int r = 0; r < 8; ++r) { mrun[r] = -INFINITY; lrun[r] = 0.f; }

    const int q0w    = q0 + w * 16;
    const int nTiles = (T_PAST + q0 + 128) / 64;

    for (int kt = 0; kt < nTiles; ++kt) {
        const int kv0 = kt * 64;
        __syncthreads();
        // K: straight b128 stores
        *(uint4*)(Ks + krow * LD + kcol)     = gk0;
        *(uint4*)(Ks + krow * LD + kcol + 8) = gk1;
        // V: transpose via half-word merges ((row2p, row2p+1) -> packed pairs)
        {
            const unsigned a0 = gva0.x, a1 = gva0.y, b0 = gvb0.x, b1 = gvb0.y;
            *(unsigned*)(Vs + (vdc + 0) * LDV + 2 * vpr) = (a0 & 0xFFFFu) | (b0 << 16);
            *(unsigned*)(Vs + (vdc + 1) * LDV + 2 * vpr) = (a0 >> 16)     | (b0 & 0xFFFF0000u);
            *(unsigned*)(Vs + (vdc + 2) * LDV + 2 * vpr) = (a1 & 0xFFFFu) | (b1 << 16);
            *(unsigned*)(Vs + (vdc + 3) * LDV + 2 * vpr) = (a1 >> 16)     | (b1 & 0xFFFF0000u);
        }
        {
            const unsigned a0 = gva1.x, a1 = gva1.y, b0 = gvb1.x, b1 = gvb1.y;
            const int pc = 2 * (vpr + 16);
            *(unsigned*)(Vs + (vdc + 0) * LDV + pc) = (a0 & 0xFFFFu) | (b0 << 16);
            *(unsigned*)(Vs + (vdc + 1) * LDV + pc) = (a0 >> 16)     | (b0 & 0xFFFF0000u);
            *(unsigned*)(Vs + (vdc + 2) * LDV + pc) = (a1 & 0xFFFFu) | (b1 << 16);
            *(unsigned*)(Vs + (vdc + 3) * LDV + pc) = (a1 >> 16)     | (b1 & 0xFFFF0000u);
        }
        __syncthreads();

        // Overlap next tile's global fetch with this tile's WMMA/softmax.
        if (kt + 1 < nTiles) {
            const size_t step = (size_t)(kv0 + 64) * 64;
            gk0  = *(const uint4*)(krp + step);
            gk1  = *(const uint4*)(krp + step + 8);
            gva0 = *(const uint2*)(vrp0 + step);
            gvb0 = *(const uint2*)(vrp0 + step + 64);
            gva1 = *(const uint2*)(vrp1 + step);
            gvb1 = *(const uint2*)(vrp1 + step + 64);
        }
        if (kt + 2 < nTiles) {   // L2 warm-up two tiles ahead (global_prefetch_b8)
            __builtin_prefetch(krp + (size_t)(kv0 + 128) * 64, 0, 3);
            __builtin_prefetch(vrp0 + (size_t)(kv0 + 128) * 64, 0, 3);
        }

        // S = Q K^T   (16x64 per wave; d=64 split into two K=32 steps)
        v8f s[4];
#pragma unroll
        for (int i = 0; i < 4; ++i)
            s[i] = (v8f){0.f, 0.f, 0.f, 0.f, 0.f, 0.f, 0.f, 0.f};
#pragma unroll
        for (int ks = 0; ks < 64; ks += 32) {
            const v16bf a = load_a_frag(Qs, LD, w * 16, ks, lane);
            v16bf bfr[4];
#pragma unroll
            for (int nt = 0; nt < 4; ++nt)
                bfr[nt] = load_b_frag_nt(Ks, LD, nt * 16, ks, lane);
#pragma unroll
            for (int nt = 0; nt < 4; ++nt)
                s[nt] = wmma_bf16(a, bfr[nt], s[nt]);
        }

        // Causal mask (uniform branch: EXEC stays full around WMMA)
        if (kv0 + 63 > T_PAST + q0w) {
#pragma unroll
            for (int nt = 0; nt < 4; ++nt) {
                const int j = kv0 + nt * 16 + ln;
#pragma unroll
                for (int r = 0; r < 8; ++r) {
                    const int i = q0w + r + hi * 8;
                    if (j > T_PAST + i) s[nt][r] = -3.0e38f;
                }
            }
        }

        // Online softmax (rows live across lanes 0-15 / 16-31 half-waves)
#pragma unroll
        for (int r = 0; r < 8; ++r) {
            float mloc = fmaxf(fmaxf(s[0][r], s[1][r]), fmaxf(s[2][r], s[3][r]));
#pragma unroll
            for (int off = 8; off >= 1; off >>= 1)
                mloc = fmaxf(mloc, __shfl_xor(mloc, off, 32));
            const float mnew  = fmaxf(mrun[r], mloc);
            const float alpha = exp2f((mrun[r] - mnew) * LOG2E);
            float rsum = 0.f;
#pragma unroll
            for (int nt = 0; nt < 4; ++nt) {
                const float p = exp2f((s[nt][r] - mnew) * LOG2E);
                s[nt][r] = p;
                rsum += p;
            }
#pragma unroll
            for (int off = 8; off >= 1; off >>= 1)
                rsum += __shfl_xor(rsum, off, 32);
            lrun[r] = lrun[r] * alpha + rsum;
            mrun[r] = mnew;
#pragma unroll
            for (int nt = 0; nt < 4; ++nt)
                accO[nt][r] *= alpha;
        }

        // Spill P to wave-private LDS region to re-fragment as WMMA A operand
#pragma unroll
        for (int nt = 0; nt < 4; ++nt)
#pragma unroll
            for (int r = 0; r < 8; ++r)
                Ps[(w * 16 + r + hi * 8) * LD + nt * 16 + ln] = f2bf(s[nt][r]);

        // O += P V   (K=64 kv positions, two K=32 steps; Vs is [d][kv] -> NT loads)
#pragma unroll
        for (int ks = 0; ks < 64; ks += 32) {
            const v16bf a = load_a_frag(Ps, LD, w * 16, ks, lane);
            v16bf bfr[4];
#pragma unroll
            for (int nt = 0; nt < 4; ++nt)
                bfr[nt] = load_b_frag_nt(Vs, LDV, nt * 16, ks, lane);
#pragma unroll
            for (int nt = 0; nt < 4; ++nt)
                accO[nt] = wmma_bf16(a, bfr[nt], accO[nt]);
        }
    }

    // Normalize; write bf16 context in [B, TQ, H*64] layout (feeds out-proj GEMM)
#pragma unroll
    for (int r = 0; r < 8; ++r) {
        const float inv = 1.0f / lrun[r];
        const int i = q0 + w * 16 + r + hi * 8;
        const size_t rowoff = ((size_t)b * 2048 + i) * 1024 + h * 64;
#pragma unroll
        for (int nt = 0; nt < 4; ++nt)
            CtxB[rowoff + nt * 16 + ln] = f2bf(accO[nt][r] * inv);
    }
}

// Copy past_K/past_V ([B,H,2048,64]) into cache slots 0..2047 of [B,H,4096,64],
// writing both the f32 output cache and the bf16 working cache.
__global__ __launch_bounds__(256)
void mha_copy_past(const float4* __restrict__ pk, const float4* __restrict__ pv,
                   float4* __restrict__ ok, float4* __restrict__ ov,
                   uint2* __restrict__ okb, uint2* __restrict__ ovb) {
    const int idx = blockIdx.x * 256 + threadIdx.x;   // 1,048,576 float4 per tensor
    const int row = idx >> 4;                         // [B*H*2048)
    const int rem = idx & 15;
    const int bh  = row >> 11;
    const int t   = row & 2047;
    const size_t o = ((size_t)bh * 4096 + t) * 16 + rem;
    const float4 a = pk[idx];
    const float4 v = pv[idx];
    ok[o] = a;
    ov[o] = v;
    union { bf16_t h[4]; uint2 u; } p;
    p.h[0] = f2bf(a.x); p.h[1] = f2bf(a.y); p.h[2] = f2bf(a.z); p.h[3] = f2bf(a.w);
    okb[o] = p.u;
    p.h[0] = f2bf(v.x); p.h[1] = f2bf(v.y); p.h[2] = f2bf(v.z); p.h[3] = f2bf(v.w);
    ovb[o] = p.u;
}

extern "C" void kernel_launch(void* const* d_in, const int* in_sizes, int n_in,
                              void* d_out, int out_size, void* d_ws, size_t ws_size,
                              hipStream_t stream) {
    (void)in_sizes; (void)n_in; (void)out_size; (void)ws_size;

    const float* query = (const float*)d_in[0];
    const float* key   = (const float*)d_in[1];
    const float* value = (const float*)d_in[2];
    const float* pastK = (const float*)d_in[3];
    const float* pastV = (const float*)d_in[4];
    // d_in[5] = mask: causality computed analytically
    const float* Wq = (const float*)d_in[6];
    const float* bq = (const float*)d_in[7];
    const float* Wk = (const float*)d_in[8];
    const float* bk = (const float*)d_in[9];
    const float* Wv = (const float*)d_in[10];
    const float* bv = (const float*)d_in[11];
    const float* Wo = (const float*)d_in[12];
    const float* bo = (const float*)d_in[13];

    float* out  = (float*)d_out;                 // [2,2048,1024]
    float* outK = out  + (size_t)4194304;        // [2,16,4096,64]
    float* outV = outK + (size_t)8388608;        // [2,16,4096,64]

    // Workspace layout (bf16 halves; ~80 MB total)
    bf16_t* p    = (bf16_t*)d_ws;
    bf16_t* Qbf  = p;  p += (size_t)4194304;     // [B,H,2048,64], pre-scaled
    bf16_t* Kbf  = p;  p += (size_t)8388608;     // [B,H,4096,64]
    bf16_t* Vbf  = p;  p += (size_t)8388608;     // [B,H,4096,64]
    bf16_t* Ctxb = p;  p += (size_t)4194304;     // [B,2048,1024]
    bf16_t* Wqb  = p;  p += (size_t)1048576;
    bf16_t* Wkb  = p;  p += (size_t)1048576;
    bf16_t* Wvb  = p;  p += (size_t)1048576;
    bf16_t* Wob  = p;  p += (size_t)1048576;
    bf16_t* Xqb  = p;  p += (size_t)4194304;
    bf16_t* Xkb  = p;  p += (size_t)4194304;
    bf16_t* Xvb  = p;  p += (size_t)4194304;

    // 0) one-time bf16 materialization (inputs + weights)
    cvt_f32_to_bf16<<<4096, 256, 0, stream>>>((const float4*)query, (uint2*)Xqb, 1048576);
    cvt_f32_to_bf16<<<4096, 256, 0, stream>>>((const float4*)key,   (uint2*)Xkb, 1048576);
    cvt_f32_to_bf16<<<4096, 256, 0, stream>>>((const float4*)value, (uint2*)Xvb, 1048576);
    cvt_f32_to_bf16<<<1024, 256, 0, stream>>>((const float4*)Wq, (uint2*)Wqb, 262144);
    cvt_f32_to_bf16<<<1024, 256, 0, stream>>>((const float4*)Wk, (uint2*)Wkb, 262144);
    cvt_f32_to_bf16<<<1024, 256, 0, stream>>>((const float4*)Wv, (uint2*)Wvb, 262144);
    cvt_f32_to_bf16<<<1024, 256, 0, stream>>>((const float4*)Wo, (uint2*)Wob, 262144);

    // 1) past KV -> f32 cache slots + bf16 working cache
    mha_copy_past<<<4096, 256, 0, stream>>>(
        (const float4*)pastK, (const float4*)pastV,
        (float4*)outK, (float4*)outV, (uint2*)Kbf, (uint2*)Vbf);

    // 2) projections (M = B*TQ = 4096, N = K = 1024)
    dim3 gg(8, 32);
    mha_gemm_wmma<<<gg, 256, 0, stream>>>(Xqb, Wqb, bq, nullptr, Qbf, 1);
    mha_gemm_wmma<<<gg, 256, 0, stream>>>(Xkb, Wkb, bk, outK, Kbf, 2);
    mha_gemm_wmma<<<gg, 256, 0, stream>>>(Xvb, Wvb, bv, outV, Vbf, 2);

    // 3) flash attention over the bf16 cache
    dim3 ga(16, 16, 2);
    mha_flash_attn_wmma<<<ga, 256, 0, stream>>>(Qbf, Kbf, Vbf, Ctxb);

    // 4) output projection (reads bf16 context, writes f32 out)
    mha_gemm_wmma<<<gg, 256, 0, stream>>>(Ctxb, Wob, bo, out, nullptr, 0);
}
